// LLM_MAIN_50998441673034
// MI455X (gfx1250) — matrix operand order repaired and checked by reading, not compiled, not run on hardware
//
#include <hip/hip_runtime.h>
#include <hip/hip_fp16.h>

typedef _Float16 half_t;
typedef unsigned int u32;
typedef __attribute__((ext_vector_type(16))) _Float16 v16h;
typedef __attribute__((ext_vector_type(8)))  float    v8f;

union FragH { v16h v; u32 u[8]; };

#define L_    2
#define B_    4
#define S_    64
#define PAST_ 2048
#define T_    (PAST_ + S_)      // 2112
#define H_    16
#define KVH_  8
#define G_    2
#define QKH_  24                // H + KVH
#define NQKV_ 32                // QKH + KVH
#define HD_   128
#define HID_  2048
#define FFN_  8192
#define V_    32000
#define M_    (B_ * S_)         // 256

static __device__ __forceinline__ v8f v8f_zero() {
    v8f z;
    #pragma unroll
    for (int i = 0; i < 8; ++i) z[i] = 0.0f;
    return z;
}

// LDS byte offset of a shared-memory pointer (low 32 bits of the generic address)
static __device__ __forceinline__ u32 lds_off(const void* p) {
    return (u32)(uintptr_t)p;
}

// ---------------------------------------------------------------------------
// RMS norm (reference: x * rsqrt(sum(x*x)) -- no epsilon, no mean), f32 -> f16
// one block per row
// ---------------------------------------------------------------------------
__global__ __launch_bounds__(256)
void rms_to_f16(const float* __restrict__ X, half_t* __restrict__ Y, int K) {
    __shared__ float red[256];
    const int row = blockIdx.x;
    const float* x = X + (size_t)row * K;
    float s = 0.0f;
    for (int k = threadIdx.x; k < K; k += 256) { float v = x[k]; s += v * v; }
    red[threadIdx.x] = s;
    __syncthreads();
    for (int off = 128; off > 0; off >>= 1) {
        if ((int)threadIdx.x < off) red[threadIdx.x] += red[threadIdx.x + off];
        __syncthreads();
    }
    const float r = rsqrtf(red[0]);
    half_t* y = Y + (size_t)row * K;
    for (int k = threadIdx.x; k < K; k += 256) y[k] = (half_t)(x[k] * r);
}

// ---------------------------------------------------------------------------
// WMMA GEMM: Y[m,n] (+)= sum_k A[m,k] * W[n,k]
//   A: f16 row-major [M x K], W: f32 row-major [N x K], Y: f32 [M x N]
//   workgroup tile 128x128, 8 waves of 64x32, K-step 32
//   A tile staged via CDNA5 async global->LDS loads (ASYNCcnt);
//   W tile converted fp32->fp16 through VALU; next W tile prefetched.
// ---------------------------------------------------------------------------
template <bool ADD>
__global__ __launch_bounds__(256)
void gemm_wmma(const half_t* __restrict__ A, const float* __restrict__ W,
               float* __restrict__ Y, int N, int K) {
    __shared__ half_t As[128 * 32];   // [row][k], row stride 32 halves (16B aligned)
    __shared__ half_t Bs[128 * 34];   // [row][k], row stride 34 halves (padded)

    const int mBase = blockIdx.x * 128;
    const int nBase = blockIdx.y * 128;
    const int tid   = threadIdx.x;
    const int lane  = tid & 31;
    const int wid   = tid >> 5;
    const int hi    = lane >> 4;
    const int lo    = lane & 15;
    const int waveM = (wid >> 2) * 64;   // 0 or 64
    const int waveN = (wid & 3) * 32;    // 0,32,64,96

    v8f acc[4][2];
    #pragma unroll
    for (int i = 0; i < 4; ++i)
        #pragma unroll
        for (int j = 0; j < 2; ++j) acc[i][j] = v8f_zero();

    const u32* AsU = (const u32*)As;
    const u32* BsU = (const u32*)Bs;

    for (int k0 = 0; k0 < K; k0 += 32) {
        // ---- stage A tile via async global->LDS (2 x 16B per thread) ----
        {
            int c = tid;
            #pragma unroll
            for (int it = 0; it < 2; ++it, c += 256) {
                const int row = c >> 2, col = (c & 3) * 8;
                const half_t* gp = A + (size_t)(mBase + row) * K + k0 + col;
                const u32 la = lds_off(&As[row * 32 + col]);
                asm volatile("global_load_async_to_lds_b128 %0, %1, off"
                             :: "v"(la), "v"(gp) : "memory");
            }
        }
        // ---- stage W tile: 128 rows x 32 floats -> f16 (4 x float4/thread) ----
        {
            int c = tid;
            #pragma unroll
            for (int it = 0; it < 4; ++it, c += 256) {
                const int row = c >> 3, col = (c & 7) * 4;
                const float4 f = *(const float4*)(W + (size_t)(nBase + row) * K + k0 + col);
                half_t* dst = Bs + row * 34 + col;
                dst[0] = (half_t)f.x; dst[1] = (half_t)f.y;
                dst[2] = (half_t)f.z; dst[3] = (half_t)f.w;
            }
        }
        // ---- prefetch next W tile (global_prefetch_b8, no counter) ----
        if (k0 + 32 < K) {
            int c = tid;
            #pragma unroll
            for (int it = 0; it < 4; ++it, c += 256) {
                const int row = c >> 3, col = (c & 7) * 4;
                __builtin_prefetch(W + (size_t)(nBase + row) * K + k0 + 32 + col, 0, 1);
            }
        }
        asm volatile("s_wait_asynccnt 0" ::: "memory");
        __syncthreads();

        // ---- fragments (ISA 7.12.2 layouts) ----
        FragH a[4], b[2];
        #pragma unroll
        for (int mf = 0; mf < 4; ++mf) {
            const int m = waveM + mf * 16 + lo;
            #pragma unroll
            for (int j = 0; j < 8; ++j) {
                const int k = (j < 4 ? 2 * j : 16 + 2 * (j - 4)) + hi * 8;  // A layout
                a[mf].u[j] = AsU[m * 16 + (k >> 1)];
            }
        }
        #pragma unroll
        for (int nf = 0; nf < 2; ++nf) {
            const int n = waveN + nf * 16 + lo;
            #pragma unroll
            for (int j = 0; j < 8; ++j)                                     // B layout
                b[nf].u[j] = BsU[n * 17 + (j < 4 ? j : j + 4) + hi * 4];
        }
        #pragma unroll
        for (int mf = 0; mf < 4; ++mf)
            #pragma unroll
            for (int nf = 0; nf < 2; ++nf)
                acc[mf][nf] = __builtin_amdgcn_wmma_f32_16x16x32_f16(
                    false, a[mf].v, false, b[nf].v, (short)0, acc[mf][nf], false, false);
        __syncthreads();
    }

    #pragma unroll
    for (int mf = 0; mf < 4; ++mf)
        #pragma unroll
        for (int nf = 0; nf < 2; ++nf)
            #pragma unroll
            for (int r = 0; r < 8; ++r) {
                const int row = mBase + waveM + mf * 16 + r + hi * 8;
                const int col = nBase + waveN + nf * 16 + lo;
                const float v = acc[mf][nf][r];
                if (ADD) Y[(size_t)row * N + col] += v;
                else     Y[(size_t)row * N + col]  = v;
            }
}

// ---------------------------------------------------------------------------
// Copy + transpose KV caches into attention-friendly layouts
//   k_cache[i]: [B,KVH,HD,PAST]  ->  k_all : [B,KVH,T,HD]   (t-major rows)
//   v_cache[i]: [B,KVH,PAST,HD]  ->  v_allT: [B,KVH,HD,T]   (hd-major rows)
// ---------------------------------------------------------------------------
__global__ void k_cache_to_all(const half_t* __restrict__ kc, half_t* __restrict__ k_all) {
    size_t idx = (size_t)blockIdx.x * blockDim.x + threadIdx.x;
    const size_t total = (size_t)B_ * KVH_ * PAST_ * HD_;
    if (idx >= total) return;
    const int hd = idx % HD_;
    const size_t r = idx / HD_;
    const int t  = r % PAST_;
    const int bk = (int)(r / PAST_);
    k_all[((size_t)bk * T_ + t) * HD_ + hd] = kc[((size_t)bk * HD_ + hd) * PAST_ + t];
}

__global__ void v_cache_to_all(const half_t* __restrict__ vc, half_t* __restrict__ v_allT) {
    size_t idx = (size_t)blockIdx.x * blockDim.x + threadIdx.x;
    const size_t total = (size_t)B_ * KVH_ * HD_ * PAST_;
    if (idx >= total) return;
    const int t = idx % PAST_;
    const size_t r = idx / PAST_;
    const int hd = r % HD_;
    const int bk = (int)(r / HD_);
    v_allT[((size_t)bk * HD_ + hd) * T_ + t] = vc[((size_t)bk * PAST_ + t) * HD_ + hd];
}

// ---------------------------------------------------------------------------
// qkv postprocess: per-head RMS * qk_norm_w, RoPE (partner = d ^ 64),
// scatter q -> q_f16 [B,KVH,G,S,HD], k -> k_all[..,PAST+s,:], v -> v_allT[..,:,PAST+s]
// one block of 128 threads per (b, s, head)
// ---------------------------------------------------------------------------
__global__ __launch_bounds__(128)
void qkv_post(const float* __restrict__ qkv, const float* __restrict__ qknw,
              const float* __restrict__ cosb, const float* __restrict__ sinb,
              half_t* __restrict__ qf, half_t* __restrict__ k_all,
              half_t* __restrict__ v_allT) {
    __shared__ float nbuf[HD_];
    __shared__ float red[128];
    const int blk  = blockIdx.x;
    const int head = blk & 31;
    const int bs   = blk >> 5;          // b*S + s
    const int s = bs % S_, b = bs / S_;
    const int d = threadIdx.x;
    const float x = qkv[((size_t)bs * NQKV_ + head) * HD_ + d];

    if (head < QKH_) {
        red[d] = x * x;
        __syncthreads();
        for (int off = 64; off > 0; off >>= 1) {
            if (d < off) red[d] += red[d + off];
            __syncthreads();
        }
        const float nv = x * rsqrtf(red[0]) * qknw[head * HD_ + d];
        nbuf[d] = nv;
        __syncthreads();
        const float o = nv * cosb[s * HD_ + d] + nbuf[d ^ 64] * sinb[s * HD_ + d];
        if (head < H_) {
            const int kvh = head >> 1, g = head & 1;   // G == 2
            qf[((((size_t)b * KVH_ + kvh) * G_ + g) * S_ + s) * HD_ + d] = (half_t)o;
        } else {
            const int kvh = head - H_;
            k_all[(((size_t)b * KVH_ + kvh) * T_ + (PAST_ + s)) * HD_ + d] = (half_t)o;
        }
    } else {
        const int kvh = head - QKH_;
        v_allT[(((size_t)b * KVH_ + kvh) * HD_ + d) * T_ + (PAST_ + s)] = (half_t)x;
    }
}

// ---------------------------------------------------------------------------
// Flash attention with WMMA. One workgroup per (b, kvh, g); 4 waves x 16 rows.
// scores = Q(64x128) @ K^T tiles of 32 cols; online softmax; O += P @ V.
// Reference has NO 1/sqrt(d) scale; causal mask: allowed iff t <= PAST + s.
// ---------------------------------------------------------------------------
__global__ __launch_bounds__(128)
void flash_attn(const half_t* __restrict__ qf, const half_t* __restrict__ k_all,
                const half_t* __restrict__ v_allT, half_t* __restrict__ of) {
    __shared__ half_t Pt[4][16][32];    // per-wave P tile re-layout buffer

    const int prob = blockIdx.x;                        // b*KVH*G + kvh*G + g
    const int g = prob % G_;
    const int kvh = (prob / G_) % KVH_;
    const int b = prob / (G_ * KVH_);
    const int head = kvh * G_ + g;
    const int wid = threadIdx.x >> 5, lane = threadIdx.x & 31;
    const int hi = lane >> 4, lo = lane & 15;
    const int sBase = wid * 16;

    const half_t* qp = qf + ((((size_t)b * KVH_ + kvh) * G_ + g) * S_ + sBase) * HD_;
    const half_t* kp = k_all + ((size_t)b * KVH_ + kvh) * (size_t)T_ * HD_;
    const half_t* vp = v_allT + ((size_t)b * KVH_ + kvh) * (size_t)HD_ * T_;

    // preload 4 Q A-fragments (K chunks of 32 along HD)
    FragH qa[4];
    {
        const u32* qrow = (const u32*)(qp + (size_t)lo * HD_);
        #pragma unroll
        for (int kc = 0; kc < 4; ++kc)
            #pragma unroll
            for (int j = 0; j < 8; ++j) {
                const int k = kc * 32 + (j < 4 ? 2 * j : 16 + 2 * (j - 4)) + hi * 8;
                qa[kc].u[j] = qrow[k >> 1];
            }
    }

    v8f oacc[8];
    #pragma unroll
    for (int c = 0; c < 8; ++c) oacc[c] = v8f_zero();
    float mrow[8], lrow[8];
    #pragma unroll
    for (int r = 0; r < 8; ++r) { mrow[r] = -1e30f; lrow[r] = 0.0f; }

    const int tmax = PAST_ + sBase + 16;   // first fully-masked tile start

    for (int t0 = 0; t0 < T_; t0 += 32) {
        if (t0 >= tmax) break;

        // prefetch next tile's K rows / V rows for this lane
        if (t0 + 32 < T_) {
            __builtin_prefetch(kp + (size_t)(t0 + 32 + lo) * HD_, 0, 1);
            __builtin_prefetch(vp + (size_t)lo * T_ + t0 + 32, 0, 1);
        }

        // ---- scores: two 16x16 fragments over 32 t-columns ----
        v8f sc[2];
        sc[0] = v8f_zero(); sc[1] = v8f_zero();
        #pragma unroll
        for (int nf = 0; nf < 2; ++nf) {
            const int t = t0 + nf * 16 + lo;
            const u32* krow = (const u32*)(kp + (size_t)t * HD_);
            #pragma unroll
            for (int kc = 0; kc < 4; ++kc) {
                FragH kb;
                #pragma unroll
                for (int j = 0; j < 8; ++j) kb.u[j] = krow[kc * 16 + (j < 4 ? j : j + 4) + hi * 4];
                sc[nf] = __builtin_amdgcn_wmma_f32_16x16x32_f16(
                    false, qa[kc].v, false, kb.v, (short)0, sc[nf], false, false);
            }
        }

        // ---- mask + online softmax (rows live in half-waves) ----
        #pragma unroll
        for (int r = 0; r < 8; ++r) {
            const int srow = sBase + r + hi * 8;
            const int ta = t0 + lo, tb = t0 + 16 + lo;
            float s0 = (ta <= PAST_ + srow) ? sc[0][r] : -1e30f;
            float s1 = (tb <= PAST_ + srow) ? sc[1][r] : -1e30f;
            float mx = fmaxf(s0, s1);
            for (int off = 1; off < 16; off <<= 1)
                mx = fmaxf(mx, __shfl_xor(mx, off, 32));
            const float mnew = fmaxf(mrow[r], mx);
            const float p0 = __expf(s0 - mnew);
            const float p1 = __expf(s1 - mnew);
            float ps = p0 + p1;
            for (int off = 1; off < 16; off <<= 1)
                ps += __shfl_xor(ps, off, 32);
            const float f = __expf(mrow[r] - mnew);
            lrow[r] = lrow[r] * f + ps;
            mrow[r] = mnew;
            #pragma unroll
            for (int c = 0; c < 8; ++c) oacc[c][r] = oacc[c][r] * f;
            Pt[wid][r + hi * 8][lo]      = (half_t)p0;
            Pt[wid][r + hi * 8][16 + lo] = (half_t)p1;
        }

        __builtin_amdgcn_wave_barrier();
        asm volatile("s_wait_dscnt 0" ::: "memory");   // in-wave LDS RAW

        // ---- reload P in A-fragment layout ----
        FragH pa;
        {
            const u32* prow = (const u32*)&Pt[wid][lo][0];
            #pragma unroll
            for (int j = 0; j < 8; ++j) {
                const int k = (j < 4 ? 2 * j : 16 + 2 * (j - 4)) + hi * 8;
                pa.u[j] = prow[k >> 1];
            }
        }
        __builtin_amdgcn_wave_barrier();

        // ---- O(16x128) += P(16x32) @ V(32x128) ----
        #pragma unroll
        for (int c = 0; c < 8; ++c) {
            FragH vb;
            const u32* vrow = (const u32*)(vp + (size_t)(c * 16 + lo) * T_ + t0);
            #pragma unroll
            for (int j = 0; j < 8; ++j) vb.u[j] = vrow[(j < 4 ? j : j + 4) + hi * 4];
            oacc[c] = __builtin_amdgcn_wmma_f32_16x16x32_f16(
                false, pa.v, false, vb.v, (short)0, oacc[c], false, false);
        }
    }

    // ---- normalize and write o as f16 [B,S,H*HD] ----
    #pragma unroll
    for (int r = 0; r < 8; ++r) {
        const int srow = sBase + r + hi * 8;
        const float inv = 1.0f / lrow[r];
        #pragma unroll
        for (int c = 0; c < 8; ++c)
            of[(((size_t)b * S_ + srow) * H_ + head) * HD_ + c * 16 + lo] =
                (half_t)(oacc[c][r] * inv);
    }
}

// ---------------------------------------------------------------------------
// silu(gate) * up, f32 [M, 2*FFN] -> f16 [M, FFN]
// ---------------------------------------------------------------------------
__global__ void silu_mul(const float* __restrict__ gu, half_t* __restrict__ out) {
    size_t idx = (size_t)blockIdx.x * blockDim.x + threadIdx.x;
    const size_t total = (size_t)M_ * FFN_;
    if (idx >= total) return;
    const int f = idx % FFN_;
    const int m = (int)(idx / FFN_);
    const float gv = gu[(size_t)m * (2 * FFN_) + f];
    const float uv = gu[(size_t)m * (2 * FFN_) + FFN_ + f];
    const float sg = gv / (1.0f + __expf(-gv));
    out[idx] = (half_t)(sg * uv);
}

// ---------------------------------------------------------------------------
// host launch
// ---------------------------------------------------------------------------
extern "C" void kernel_launch(void* const* d_in, const int* in_sizes, int n_in,
                              void* d_out, int out_size, void* d_ws, size_t ws_size,
                              hipStream_t stream) {
    (void)in_sizes; (void)n_in; (void)out_size; (void)ws_size;

    const half_t* k_cache   = (const half_t*)d_in[0];   // [L,B,KVH,1,HD,PAST] f16
    const half_t* v_cache   = (const half_t*)d_in[1];   // [L,B,KVH,1,PAST,HD] f16
    const float*  hidden    = (const float*)d_in[2];    // [B,S,HID]
    const float*  cosb      = (const float*)d_in[3];    // [1,S,1,1,HD]
    const float*  sinb      = (const float*)d_in[4];    // [1,S,1,1,HD]
    /* d_in[5] = attention_mask (mask computed inline) */
    const float*  qkv_w     = (const float*)d_in[6];    // [L, 32*HD, HID]
    const float*  qk_norm_w = (const float*)d_in[7];    // [L, QKH, HD]
    const float*  o_w       = (const float*)d_in[8];    // [L, HID, H*HD]
    const float*  gate_up_w = (const float*)d_in[9];    // [L, 2*FFN, HID]
    const float*  down_w    = (const float*)d_in[10];   // [L, HID, FFN]
    const float*  lm_head_w = (const float*)d_in[11];   // [V, HID]
    float* out = (float*)d_out;                          // [B,S,V]

    // carve workspace (~65 MB)
    char* ws = (char*)d_ws;
    size_t off = 0;
    auto alloc = [&](size_t bytes) -> char* {
        char* p = ws + off;
        off = (off + bytes + 255) & ~(size_t)255;
        return p;
    };
    float*  h      = (float*) alloc((size_t)M_ * HID_ * 4);
    half_t* xn     = (half_t*)alloc((size_t)M_ * HID_ * 2);
    float*  qkv    = (float*) alloc((size_t)M_ * NQKV_ * HD_ * 4);
    half_t* qf     = (half_t*)alloc((size_t)B_ * KVH_ * G_ * S_ * HD_ * 2);
    half_t* k_all  = (half_t*)alloc((size_t)B_ * KVH_ * T_ * HD_ * 2);
    half_t* v_allT = (half_t*)alloc((size_t)B_ * KVH_ * HD_ * T_ * 2);
    half_t* of     = (half_t*)alloc((size_t)M_ * H_ * HD_ * 2);
    float*  gu     = (float*) alloc((size_t)M_ * 2 * FFN_ * 4);
    half_t* guh    = (half_t*)alloc((size_t)M_ * FFN_ * 2);

    hipMemcpyAsync(h, hidden, (size_t)M_ * HID_ * 4, hipMemcpyDeviceToDevice, stream);

    const size_t cacheElems = (size_t)B_ * KVH_ * PAST_ * HD_;  // per layer
    const int    cacheBlks  = (int)((cacheElems + 255) / 256);

    for (int i = 0; i < L_; ++i) {
        // attention block
        rms_to_f16<<<M_, 256, 0, stream>>>(h, xn, HID_);
        gemm_wmma<false><<<dim3(M_ / 128, (NQKV_ * HD_) / 128), 256, 0, stream>>>(
            xn, qkv_w + (size_t)i * NQKV_ * HD_ * HID_, qkv, NQKV_ * HD_, HID_);

        k_cache_to_all<<<cacheBlks, 256, 0, stream>>>(k_cache + (size_t)i * cacheElems, k_all);
        v_cache_to_all<<<cacheBlks, 256, 0, stream>>>(v_cache + (size_t)i * cacheElems, v_allT);

        qkv_post<<<M_ * NQKV_, 128, 0, stream>>>(
            qkv, qk_norm_w + (size_t)i * QKH_ * HD_, cosb, sinb, qf, k_all, v_allT);

        flash_attn<<<B_ * KVH_ * G_, 128, 0, stream>>>(qf, k_all, v_allT, of);

        gemm_wmma<true><<<dim3(M_ / 128, HID_ / 128), 256, 0, stream>>>(
            of, o_w + (size_t)i * HID_ * (H_ * HD_), h, HID_, H_ * HD_);

        // MLP block
        rms_to_f16<<<M_, 256, 0, stream>>>(h, xn, HID_);
        gemm_wmma<false><<<dim3(M_ / 128, (2 * FFN_) / 128), 256, 0, stream>>>(
            xn, gate_up_w + (size_t)i * 2 * FFN_ * HID_, gu, 2 * FFN_, HID_);

        {
            const size_t n = (size_t)M_ * FFN_;
            silu_mul<<<(int)((n + 255) / 256), 256, 0, stream>>>(gu, guh);
        }
        gemm_wmma<true><<<dim3(M_ / 128, HID_ / 128), 256, 0, stream>>>(
            guh, down_w + (size_t)i * HID_ * FFN_, h, HID_, FFN_);
    }

    // lm head: N = 32000 = 250 * 128
    rms_to_f16<<<M_, 256, 0, stream>>>(h, xn, HID_);
    gemm_wmma<false><<<dim3(M_ / 128, V_ / 128), 256, 0, stream>>>(
        xn, lm_head_w, out, V_, HID_);
}
